// MambaSlotAttention_25563645346594
// MI455X (gfx1250) — matrix operand order repaired and checked by reading, not compiled, not run on hardware
//
#include <hip/hip_runtime.h>
#include <hip/hip_bf16.h>
#include <math.h>

// ---------------------------------------------------------------------------
// MambaSlotAttention forward for gfx1250 (MI455X, wave32, WMMA).
// Big GEMMs -> v_wmma_f32_16x16x32_f16 (f32 accumulate), selective scan keeps
// h[64] state in VGPRs per lane, wave32 shuffle reductions for LN/l2norm.
// Input flattening assumption: recursive insertion-order of setup_inputs dict:
//   0 features, 1 slots_init, then per block b (base 2+13*b):
//   Win,conv_w,conv_b,Wx,Wdt,bdt,A_log,D,Wout,sw,sb,ng,nb ; then
//   41 ni_g,42 ni_b,43 ns_g,44 ns_b,45 Wq,46 Wk(unused),47 Wv,
//   48 W_ih,49 W_hh,50 b_ih,51 b_hh,52 mlp.g,53 mlp.b,54 W1,55 b1,56 W2,57 b2
// ---------------------------------------------------------------------------

typedef __attribute__((ext_vector_type(16))) _Float16 v16h;
typedef __attribute__((ext_vector_type(8)))  _Float16 v8h;
typedef __attribute__((ext_vector_type(4)))  _Float16 v4h;
typedef __attribute__((ext_vector_type(8)))  float    v8f;

#define D_MODEL 768
#define D_STATE 64
#define D_INNER 1536
#define DT_RANK 48
#define DBC_W   (DT_RANK + 2 * D_STATE)   // 176
#define K_SLOTS 12
#define N_ITERS 3
#define BB      4
#define NPIX    576                        // 24*24
#define EPS_ATTN 1e-8f

__device__ __forceinline__ float sigf(float x)  { return 1.f / (1.f + __expf(-x)); }
__device__ __forceinline__ float siluf(float x) { return x * sigf(x); }
__device__ __forceinline__ float softplusf(float x) { return (x > 20.f) ? x : log1pf(expf(x)); }
__device__ __forceinline__ float geluf(float x) { return 0.5f * x * (1.f + erff(x * 0.7071067811865476f)); }

// ---------------------------------------------------------------------------
// Generic WMMA GEMM: C[M,N] = act(A[M,K] @ B + bias) + residual
// A row-major (lda). TRANSB=0: B is KxN (ldb). TRANSB=1: B is NxK (ldb).
// ALIGNED=1: M%64==0 && N%64==0 && K%32==0 -> branch-free float4 staging.
// All staging loops have FIXED trip counts (128 threads) so no exec masking,
// and are split load-phase / store-phase so global loads batch under one wait.
// act: 0=none, 1=softplus, 2=gelu
// ---------------------------------------------------------------------------
#define BM 64
#define BN 64
#define BK 32

template <int TRANSB, int ALIGNED>
__global__ void __launch_bounds__(128)
gemm_wmma_kernel(const float* __restrict__ A, int lda,
                 const float* __restrict__ Bm, int ldb,
                 const float* __restrict__ bias,
                 const float* __restrict__ residual,
                 float* __restrict__ C, int ldc,
                 int M, int N, int K, int act) {
  __shared__ _Float16 sA[BM * BK];    // [m][k]
  __shared__ _Float16 sBt[BN * BK];   // [n][k]
  const int tid  = threadIdx.x & 127;
  const int lane = tid & 31;
  const int wave = tid >> 5;
  const int wy = wave >> 1, wx = wave & 1;
  const int rowBase = blockIdx.y * BM;
  const int colBase = blockIdx.x * BN;
  const int lrow = lane & 15;
  const int hi   = lane >> 4;

  v8f zf = {0.f, 0.f, 0.f, 0.f, 0.f, 0.f, 0.f, 0.f};
  v8f acc[2][2];
  acc[0][0] = zf; acc[0][1] = zf; acc[1][0] = zf; acc[1][1] = zf;

  for (int k0 = 0; k0 < K; k0 += BK) {
    // ---- prefetch next K tile (global_prefetch_b8) --------------------
    if (k0 + BK < K) {
      int pr = rowBase + (tid & 63); if (pr > M - 1) pr = M - 1;
      __builtin_prefetch(&A[(size_t)pr * lda + k0 + BK], 0, 1);
      if (TRANSB) {
        int pn = colBase + (tid & 63); if (pn > N - 1) pn = N - 1;
        __builtin_prefetch(&Bm[(size_t)pn * ldb + k0 + BK], 0, 1);
      } else {
        int pk = k0 + BK + (tid >> 2); if (pk > K - 1) pk = K - 1;
        __builtin_prefetch(&Bm[(size_t)pk * ldb + colBase], 0, 1);
      }
    }

    if (ALIGNED) {
      // ---- load phase: 4 A-float4 + 4 B-float4 per thread, batched ----
      float4 abuf[4], bbuf[4];
#pragma unroll
      for (int it = 0; it < 4; ++it) {
        int i = tid + it * 128;                 // i < 512
        int r = i >> 3, k4 = (i & 7) << 2;
        abuf[it] = *(const float4*)&A[(size_t)(rowBase + r) * lda + k0 + k4];
      }
      if (TRANSB) {
#pragma unroll
        for (int it = 0; it < 4; ++it) {
          int i = tid + it * 128;
          int n = i >> 3, k4 = (i & 7) << 2;
          bbuf[it] = *(const float4*)&Bm[(size_t)(colBase + n) * ldb + k0 + k4];
        }
      } else {
#pragma unroll
        for (int it = 0; it < 4; ++it) {
          int i = tid + it * 128;
          int kk = i >> 4, n4 = (i & 15) << 2;
          bbuf[it] = *(const float4*)&Bm[(size_t)(k0 + kk) * ldb + colBase + n4];
        }
      }
      // ---- store phase: convert + LDS store ---------------------------
#pragma unroll
      for (int it = 0; it < 4; ++it) {
        int i = tid + it * 128;
        int r = i >> 3, k4 = (i & 7) << 2;
        v4h hv = { (_Float16)abuf[it].x, (_Float16)abuf[it].y,
                   (_Float16)abuf[it].z, (_Float16)abuf[it].w };
        *(v4h*)&sA[r * BK + k4] = hv;
      }
      if (TRANSB) {
#pragma unroll
        for (int it = 0; it < 4; ++it) {
          int i = tid + it * 128;
          int n = i >> 3, k4 = (i & 7) << 2;
          v4h hv = { (_Float16)bbuf[it].x, (_Float16)bbuf[it].y,
                     (_Float16)bbuf[it].z, (_Float16)bbuf[it].w };
          *(v4h*)&sBt[n * BK + k4] = hv;
        }
      } else {
#pragma unroll
        for (int it = 0; it < 4; ++it) {
          int i = tid + it * 128;
          int kk = i >> 4, n4 = (i & 15) << 2;
          sBt[(n4 + 0) * BK + kk] = (_Float16)bbuf[it].x;
          sBt[(n4 + 1) * BK + kk] = (_Float16)bbuf[it].y;
          sBt[(n4 + 2) * BK + kk] = (_Float16)bbuf[it].z;
          sBt[(n4 + 3) * BK + kk] = (_Float16)bbuf[it].w;
        }
      }
    } else {
      // ---- unaligned: clamped unconditional loads, batched ------------
      float av[16], bv[16];
#pragma unroll
      for (int it = 0; it < 16; ++it) {
        int i = tid + it * 128;                 // i < 2048
        int r = i >> 5, kk = i & 31;
        int gm = rowBase + r, gk = k0 + kk;
        bool ok = (gm < M) && (gk < K);
        av[it] = A[(size_t)(ok ? gm : 0) * lda + (ok ? gk : 0)];
      }
#pragma unroll
      for (int it = 0; it < 16; ++it) {
        int i = tid + it * 128;
        if (TRANSB) {
          int n = i >> 5, kk = i & 31;
          int gn = colBase + n, gk = k0 + kk;
          bool ok = (gn < N) && (gk < K);
          bv[it] = Bm[(size_t)(ok ? gn : 0) * ldb + (ok ? gk : 0)];
        } else {
          int kk = i >> 6, n = i & 63;
          int gk = k0 + kk, gn = colBase + n;
          bool ok = (gk < K) && (gn < N);
          bv[it] = Bm[(size_t)(ok ? gk : 0) * ldb + (ok ? gn : 0)];
        }
      }
#pragma unroll
      for (int it = 0; it < 16; ++it) {
        int i = tid + it * 128;
        int r = i >> 5, kk = i & 31;
        int gm = rowBase + r, gk = k0 + kk;
        bool ok = (gm < M) && (gk < K);
        sA[r * BK + kk] = (_Float16)(ok ? av[it] : 0.f);
      }
#pragma unroll
      for (int it = 0; it < 16; ++it) {
        int i = tid + it * 128;
        if (TRANSB) {
          int n = i >> 5, kk = i & 31;
          int gn = colBase + n, gk = k0 + kk;
          bool ok = (gn < N) && (gk < K);
          sBt[n * BK + kk] = (_Float16)(ok ? bv[it] : 0.f);
        } else {
          int kk = i >> 6, n = i & 63;
          int gk = k0 + kk, gn = colBase + n;
          bool ok = (gk < K) && (gn < N);
          sBt[n * BK + kk] = (_Float16)(ok ? bv[it] : 0.f);
        }
      }
    }
    __syncthreads();

    // ---- fragments (contiguous LDS reads) -----------------------------
    v16h afrag[2], bfrag[2];
#pragma unroll
    for (int mi = 0; mi < 2; ++mi) {
      const _Float16* ap = &sA[(wy * 32 + mi * 16 + lrow) * BK];
      // A frag K map: k = (e>>3)*16 + hi*8 + (e&7) -> two contiguous chunks
      v8h lo8 = *(const v8h*)(ap + hi * 8);
      v8h hi8 = *(const v8h*)(ap + 16 + hi * 8);
      afrag[mi] = __builtin_shufflevector(lo8, hi8, 0, 1, 2, 3, 4, 5, 6, 7,
                                          8, 9, 10, 11, 12, 13, 14, 15);
    }
#pragma unroll
    for (int ni = 0; ni < 2; ++ni) {
      // B frag K map: k = (e>>3)*16 + hi*8 + (e&7) -> two contiguous chunks in sBt[n][k]
      const _Float16* bp = &sBt[(wx * 32 + ni * 16 + lrow) * BK];
      v8h blo8 = *(const v8h*)(bp + hi * 8);
      v8h bhi8 = *(const v8h*)(bp + 16 + hi * 8);
      bfrag[ni] = __builtin_shufflevector(blo8, bhi8, 0, 1, 2, 3, 4, 5, 6, 7,
                                          8, 9, 10, 11, 12, 13, 14, 15);
    }
#pragma unroll
    for (int mi = 0; mi < 2; ++mi)
#pragma unroll
      for (int ni = 0; ni < 2; ++ni)
        acc[mi][ni] = __builtin_amdgcn_wmma_f32_16x16x32_f16(
            false, afrag[mi], false, bfrag[ni], (short)0, acc[mi][ni], false, false);
    __syncthreads();
  }

#pragma unroll
  for (int mi = 0; mi < 2; ++mi)
#pragma unroll
    for (int ni = 0; ni < 2; ++ni)
#pragma unroll
      for (int r = 0; r < 8; ++r) {
        int m = rowBase + wy * 32 + mi * 16 + r + (hi << 3);
        int n = colBase + wx * 32 + ni * 16 + lrow;
        if (m < M && n < N) {
          float x = acc[mi][ni][r];
          if (bias) x += bias[n];
          if (act == 1) x = softplusf(x);
          else if (act == 2) x = geluf(x);
          if (residual) x += residual[(size_t)m * ldc + n];
          C[(size_t)m * ldc + n] = x;
        }
      }
}

// ---------------------------------------------------------------------------
// LayerNorm (one wave32 per row, D == 768) with optional l2norm second output
// ---------------------------------------------------------------------------
__global__ void ln_kernel(const float* __restrict__ in, const float* __restrict__ g,
                          const float* __restrict__ bb, float* __restrict__ out,
                          float* __restrict__ out_l2, int M, int D) {
  int wid = threadIdx.x >> 5, lane = threadIdx.x & 31;
  int row = blockIdx.x * (blockDim.x >> 5) + wid;
  if (row >= M) return;
  const float* x = in + (size_t)row * D;
  float vals[24];
  float s = 0.f, sq = 0.f;
  int cnt = 0;
  for (int d = lane; d < D; d += 32) { float t = x[d]; vals[cnt++] = t; s += t; sq += t * t; }
  for (int m = 16; m >= 1; m >>= 1) { s += __shfl_xor(s, m, 32); sq += __shfl_xor(sq, m, 32); }
  float mean = s / D;
  float var  = sq / D - mean * mean;
  float rstd = rsqrtf(var + 1e-5f);
  float l2 = 0.f;
  cnt = 0;
  for (int d = lane; d < D; d += 32) {
    float y = (vals[cnt] - mean) * rstd * g[d] + bb[d];
    vals[cnt++] = y;
    out[(size_t)row * D + d] = y;
    l2 += y * y;
  }
  if (out_l2) {
    for (int m = 16; m >= 1; m >>= 1) l2 += __shfl_xor(l2, m, 32);
    float inv = 1.f / fmaxf(sqrtf(l2), 1e-12f);
    cnt = 0;
    for (int d = lane; d < D; d += 32) out_l2[(size_t)row * D + d] = vals[cnt++] * inv;
  }
}

__global__ void l2norm_kernel(const float* __restrict__ in, float* __restrict__ out,
                              int M, int D) {
  int wid = threadIdx.x >> 5, lane = threadIdx.x & 31;
  int row = blockIdx.x * (blockDim.x >> 5) + wid;
  if (row >= M) return;
  const float* x = in + (size_t)row * D;
  float l2 = 0.f;
  for (int d = lane; d < D; d += 32) { float t = x[d]; l2 += t * t; }
  for (int m = 16; m >= 1; m >>= 1) l2 += __shfl_xor(l2, m, 32);
  float inv = 1.f / fmaxf(sqrtf(l2), 1e-12f);
  for (int d = lane; d < D; d += 32) out[(size_t)row * D + d] = x[d] * inv;
}

// ---------------------------------------------------------------------------
// Causal depthwise conv1d (taps=4) + SiLU.  Input xc lives in xz (ld 2*D_INNER)
// ---------------------------------------------------------------------------
__global__ void dwconv1d_silu_kernel(const float* __restrict__ xz, int ld_in,
                                     const float* __restrict__ w, const float* __restrict__ bias,
                                     float* __restrict__ out, int Bb, int L, int DI) {
  long idx = (long)blockIdx.x * blockDim.x + threadIdx.x;
  long total = (long)Bb * L * DI;
  if (idx >= total) return;
  int c = (int)(idx % DI);
  int t = (int)((idx / DI) % L);
  int b = (int)(idx / ((long)DI * L));
  float acc = bias[c];
#pragma unroll
  for (int j = 0; j < 4; ++j) {
    int tt = t - 3 + j;
    if (tt >= 0) acc += w[c * 4 + j] * xz[(size_t)(b * L + tt) * ld_in + c];
  }
  out[idx] = siluf(acc);
}

// ---------------------------------------------------------------------------
// Selective scan. One thread per (b, channel), h[64] in VGPRs, B/C per t in LDS
// Fused epilogue: (y + D*x) * silu(z)
// ---------------------------------------------------------------------------
__global__ void __launch_bounds__(256)
scan_kernel(const float* __restrict__ delta, const float* __restrict__ xin,
            const float* __restrict__ Bmat, int ldB,
            const float* __restrict__ Cmat, int ldC,
            const float* __restrict__ zin, int ldz,
            const float* __restrict__ A_log, const float* __restrict__ Dp,
            float* __restrict__ out, int Bb, int L, int DI, int blocksPerB) {
  __shared__ float sB[D_STATE], sC[D_STATE], sA[D_STATE];
  int b = blockIdx.x / blocksPerB;
  int c = (blockIdx.x % blocksPerB) * blockDim.x + threadIdx.x;
  if (threadIdx.x < D_STATE) sA[threadIdx.x] = -__expf(A_log[threadIdx.x]);
  float h[D_STATE];
#pragma unroll
  for (int n = 0; n < D_STATE; ++n) h[n] = 0.f;
  float Dc = (c < DI) ? Dp[c] : 0.f;
  for (int t = 0; t < L; ++t) {
    __syncthreads();
    int row = b * L + t;
    if (threadIdx.x < D_STATE)
      sB[threadIdx.x] = Bmat[(size_t)row * ldB + threadIdx.x];
    else if (threadIdx.x < 2 * D_STATE)
      sC[threadIdx.x - D_STATE] = Cmat[(size_t)row * ldC + (threadIdx.x - D_STATE)];
    __syncthreads();
    if (c < DI) {
      float dt = delta[(size_t)row * DI + c];
      float xt = xin[(size_t)row * DI + c];
      float dx = dt * xt;
      float y = 0.f;
#pragma unroll
      for (int n = 0; n < D_STATE; ++n) {
        h[n] = __expf(dt * sA[n]) * h[n] + dx * sB[n];
        y += h[n] * sC[n];
      }
      float zt = zin[(size_t)row * ldz + c];
      out[(size_t)row * DI + c] = (y + Dc * xt) * siluf(zt);
    }
  }
}

// ---------------------------------------------------------------------------
// 3x3 depthwise spatial conv on 24x24 grid, fused residual add: out = xs + conv
// ---------------------------------------------------------------------------
__global__ void dwconv2d_add_kernel(const float* __restrict__ xs, const float* __restrict__ w,
                                    const float* __restrict__ bias, float* __restrict__ out,
                                    int Bb) {
  long idx = (long)blockIdx.x * blockDim.x + threadIdx.x;
  long total = (long)Bb * NPIX * D_MODEL;
  if (idx >= total) return;
  int c = (int)(idx % D_MODEL);
  int x = (int)((idx / D_MODEL) % 24);
  int y = (int)((idx / D_MODEL / 24) % 24);
  int b = (int)(idx / ((long)D_MODEL * NPIX));
  float acc = bias[c];
#pragma unroll
  for (int dy = 0; dy < 3; ++dy)
#pragma unroll
    for (int dx = 0; dx < 3; ++dx) {
      int yy = y + dy - 1, xx = x + dx - 1;
      if (yy >= 0 && yy < 24 && xx >= 0 && xx < 24)
        acc += w[c * 9 + dy * 3 + dx] * xs[((size_t)(b * NPIX) + yy * 24 + xx) * D_MODEL + c];
    }
  out[idx] = xs[idx] + acc;
}

// ---------------------------------------------------------------------------
// Attention: logits = fn @ slots_norm^T * scale; softmax over K=12 (+eps);
// writes attn to d_out and accumulates per-(b,k) column sums (atomics).
// ---------------------------------------------------------------------------
__global__ void attn_kernel(const float* __restrict__ fn, const float* __restrict__ sn,
                            float* __restrict__ attn_out, float* __restrict__ colsum,
                            int Bb, int Npix, int D, int nblk, float scale) {
  __shared__ float sS[K_SLOTS * D_MODEL];
  int b = blockIdx.x / nblk;
  int n = (blockIdx.x % nblk) * blockDim.x + threadIdx.x;
  for (int i = threadIdx.x; i < K_SLOTS * D; i += blockDim.x)
    sS[i] = sn[(size_t)b * K_SLOTS * D + i];
  __syncthreads();
  if (n >= Npix) return;
  float acc[K_SLOTS];
#pragma unroll
  for (int k = 0; k < K_SLOTS; ++k) acc[k] = 0.f;
  const float* fr = fn + ((size_t)b * Npix + n) * D;
  for (int d = 0; d < D; ++d) {
    float fv = fr[d];
#pragma unroll
    for (int k = 0; k < K_SLOTS; ++k) acc[k] += fv * sS[k * D_MODEL + d];
  }
  float mx = -1e30f;
#pragma unroll
  for (int k = 0; k < K_SLOTS; ++k) { acc[k] *= scale; mx = fmaxf(mx, acc[k]); }
  float sum = 0.f;
#pragma unroll
  for (int k = 0; k < K_SLOTS; ++k) { acc[k] = __expf(acc[k] - mx); sum += acc[k]; }
  float inv = 1.f / sum;
  size_t base = ((size_t)b * Npix + n) * K_SLOTS;
#pragma unroll
  for (int k = 0; k < K_SLOTS; ++k) {
    float a = acc[k] * inv + EPS_ATTN;
    attn_out[base + k] = a;
    atomicAdd(&colsum[b * K_SLOTS + k], a);
  }
}

// updates[b,k,d] = sum_n attn[b,n,k] * v[b,n,d] / colsum[b,k]
__global__ void updates_kernel(const float* __restrict__ attn, const float* __restrict__ colsum,
                               const float* __restrict__ v, float* __restrict__ upd,
                               int Bb, int Npix, int D) {
  int chunks = D / 256;
  int ch = blockIdx.x % chunks;
  int bk = blockIdx.x / chunks;
  int b = bk / K_SLOTS, k = bk % K_SLOTS;
  int d = ch * 256 + threadIdx.x;
  float inv = 1.f / colsum[bk];
  float acc = 0.f;
  for (int n = 0; n < Npix; ++n)
    acc += attn[((size_t)b * Npix + n) * K_SLOTS + k] * v[((size_t)b * Npix + n) * D + d];
  upd[(size_t)bk * D + d] = acc * inv;
}

__global__ void gru_kernel(const float* __restrict__ gi, const float* __restrict__ gh,
                           const float* __restrict__ hprev, float* __restrict__ out,
                           int M, int D) {
  long idx = (long)blockIdx.x * blockDim.x + threadIdx.x;
  if (idx >= (long)M * D) return;
  int row = (int)(idx / D), d = (int)(idx % D);
  size_t b3 = (size_t)row * 3 * D;
  float ir = gi[b3 + d], iz = gi[b3 + D + d], in = gi[b3 + 2 * D + d];
  float hr = gh[b3 + d], hz = gh[b3 + D + d], hn = gh[b3 + 2 * D + d];
  float r = sigf(ir + hr);
  float z = sigf(iz + hz);
  float n = tanhf(in + r * hn);
  float h = hprev[idx];
  out[idx] = (1.f - z) * n + z * h;
}

__global__ void zero_kernel(float* p, int n) {
  int i = blockIdx.x * blockDim.x + threadIdx.x;
  if (i < n) p[i] = 0.f;
}

// ---------------------------------------------------------------------------
// Host-side helpers
// ---------------------------------------------------------------------------
static void gemm(hipStream_t s, const float* A, int lda, const float* Bm, int ldb, int transB,
                 const float* bias, const float* residual, float* C, int ldc,
                 int M, int N, int K, int act) {
  dim3 grid((N + BN - 1) / BN, (M + BM - 1) / BM);
  bool al = (M % BM == 0) && (N % BN == 0) && (K % BK == 0);
  if (transB) {
    if (al) gemm_wmma_kernel<1, 1><<<grid, 128, 0, s>>>(A, lda, Bm, ldb, bias, residual, C, ldc, M, N, K, act);
    else    gemm_wmma_kernel<1, 0><<<grid, 128, 0, s>>>(A, lda, Bm, ldb, bias, residual, C, ldc, M, N, K, act);
  } else {
    if (al) gemm_wmma_kernel<0, 1><<<grid, 128, 0, s>>>(A, lda, Bm, ldb, bias, residual, C, ldc, M, N, K, act);
    else    gemm_wmma_kernel<0, 0><<<grid, 128, 0, s>>>(A, lda, Bm, ldb, bias, residual, C, ldc, M, N, K, act);
  }
}

static void run_mamba(hipStream_t s, const float* x, int Bb, int L,
                      const float* Win, const float* conv_w, const float* conv_b,
                      const float* Wx, const float* Wdt, const float* bdt,
                      const float* A_log, const float* Dp, const float* Wout,
                      float* xz, float* xcact, float* dbc, float* delta, float* ys,
                      float* out) {
  int M = Bb * L;
  gemm(s, x, D_MODEL, Win, 2 * D_INNER, 0, nullptr, nullptr, xz, 2 * D_INNER, M, 2 * D_INNER, D_MODEL, 0);
  long total = (long)M * D_INNER;
  dwconv1d_silu_kernel<<<(int)((total + 255) / 256), 256, 0, s>>>(xz, 2 * D_INNER, conv_w, conv_b, xcact, Bb, L, D_INNER);
  gemm(s, xcact, D_INNER, Wx, DBC_W, 0, nullptr, nullptr, dbc, DBC_W, M, DBC_W, D_INNER, 0);
  gemm(s, dbc, DBC_W, Wdt, D_INNER, 0, bdt, nullptr, delta, D_INNER, M, D_INNER, DT_RANK, 1 /*softplus*/);
  int bpb = D_INNER / 256;
  scan_kernel<<<Bb * bpb, 256, 0, s>>>(delta, xcact, dbc + DT_RANK, DBC_W,
                                       dbc + DT_RANK + D_STATE, DBC_W,
                                       xz + D_INNER, 2 * D_INNER,
                                       A_log, Dp, ys, Bb, L, D_INNER, bpb);
  gemm(s, ys, D_INNER, Wout, D_MODEL, 0, nullptr, nullptr, out, D_MODEL, M, D_MODEL, D_INNER, 0);
}

extern "C" void kernel_launch(void* const* d_in, const int* in_sizes, int n_in,
                              void* d_out, int out_size, void* d_ws, size_t ws_size,
                              hipStream_t stream) {
  (void)in_sizes; (void)n_in; (void)out_size; (void)ws_size;
  auto F = [&](int i) -> const float* { return (const float*)d_in[i]; };
  const float* features   = F(0);
  const float* slots_init = F(1);
  const int BLK = 13;
  auto blk = [&](int b, int j) -> const float* { return F(2 + b * BLK + j); };
  const float* ni_g = F(41); const float* ni_b = F(42);
  const float* ns_g = F(43); const float* ns_b = F(44);
  const float* Wq = F(45);  /* F(46)=Wk is dead code in the reference */
  const float* Wv = F(47);
  const float* W_ih = F(48); const float* W_hh = F(49);
  const float* b_ih = F(50); const float* b_hh = F(51);
  const float* mg = F(52); const float* mbv = F(53);
  const float* W1 = F(54); const float* b1 = F(55);
  const float* W2 = F(56); const float* b2 = F(57);

  const int M  = BB * NPIX;     // 2304
  const int MS = BB * K_SLOTS;  // 48
  const float SCALE = 0.03608439182435161f;  // 768^-0.5

  float* w = (float*)d_ws;
  auto take = [&](size_t n) { float* p = w; w += n; return p; };
  float* f      = take((size_t)M * D_MODEL);
  float* fn     = take((size_t)M * D_MODEL);
  float* v      = take((size_t)M * D_MODEL);
  float* xz     = take((size_t)M * 2 * D_INNER);
  float* xcact  = take((size_t)M * D_INNER);
  float* dbc    = take((size_t)M * DBC_W);
  float* delta  = take((size_t)M * D_INNER);
  float* ys     = take((size_t)M * D_INNER);
  float* xs     = take((size_t)M * D_MODEL);
  float* tsp    = take((size_t)M * D_MODEL);
  float* slotsA = take((size_t)MS * D_MODEL);
  float* slotsB = take((size_t)MS * D_MODEL);
  float* qln    = take((size_t)MS * D_MODEL);
  float* q      = take((size_t)MS * D_MODEL);
  float* xz_s   = take((size_t)MS * 2 * D_INNER);
  float* xca_s  = take((size_t)MS * D_INNER);
  float* dbc_s  = take((size_t)MS * DBC_W);
  float* del_s  = take((size_t)MS * D_INNER);
  float* ys_s   = take((size_t)MS * D_INNER);
  float* sn     = take((size_t)MS * D_MODEL);
  float* snn    = take((size_t)MS * D_MODEL);
  float* upd    = take((size_t)MS * D_MODEL);
  float* gi     = take((size_t)MS * 3 * D_MODEL);
  float* gh     = take((size_t)MS * 3 * D_MODEL);
  float* sbuf   = take((size_t)MS * D_MODEL);
  float* tln    = take((size_t)MS * D_MODEL);
  float* h1     = take((size_t)MS * 4 * D_MODEL);
  float* colsum = take(64);

  float* out_slots = (float*)d_out;
  float* out_attn  = out_slots + (size_t)MS * D_MODEL;
  float* out_pf    = out_attn + (size_t)BB * NPIX * K_SLOTS;

  // f = LN(features); fn = l2norm(f)
  ln_kernel<<<(M + 7) / 8, 256, 0, stream>>>(features, ni_g, ni_b, f, fn, M, D_MODEL);
  // v = f @ Wv
  gemm(stream, f, D_MODEL, Wv, D_MODEL, 0, nullptr, nullptr, v, D_MODEL, M, D_MODEL, D_MODEL, 0);

  // pf = _spatial(f, blocks[2]) — only the last iteration's pf is returned,
  // and f is loop-invariant, so compute the heavy spatial path exactly once.
  {
    int bi = N_ITERS - 1;
    run_mamba(stream, f, BB, NPIX,
              blk(bi, 0), blk(bi, 1), blk(bi, 2), blk(bi, 3), blk(bi, 4),
              blk(bi, 5), blk(bi, 6), blk(bi, 7), blk(bi, 8),
              xz, xcact, dbc, delta, ys, xs);
    long total = (long)M * D_MODEL;
    dwconv2d_add_kernel<<<(int)((total + 255) / 256), 256, 0, stream>>>(xs, blk(bi, 9), blk(bi, 10), tsp, BB);
    ln_kernel<<<(M + 7) / 8, 256, 0, stream>>>(tsp, blk(bi, 11), blk(bi, 12), out_pf, nullptr, M, D_MODEL);
  }

  hipMemcpyAsync(slotsA, slots_init, (size_t)MS * D_MODEL * sizeof(float),
                 hipMemcpyDeviceToDevice, stream);
  float* scur = slotsA;
  float* snxt = slotsB;

  for (int it = 0; it < N_ITERS; ++it) {
    // q = LN(slots) @ Wq
    ln_kernel<<<(MS + 7) / 8, 256, 0, stream>>>(scur, ns_g, ns_b, qln, nullptr, MS, D_MODEL);
    gemm(stream, qln, D_MODEL, Wq, D_MODEL, 0, nullptr, nullptr, q, D_MODEL, MS, D_MODEL, D_MODEL, 0);
    // slots_new = mamba(q, blocks[0].mamba), L = 12
    run_mamba(stream, q, BB, K_SLOTS,
              blk(0, 0), blk(0, 1), blk(0, 2), blk(0, 3), blk(0, 4),
              blk(0, 5), blk(0, 6), blk(0, 7), blk(0, 8),
              xz_s, xca_s, dbc_s, del_s, ys_s, sn);
    l2norm_kernel<<<(MS + 7) / 8, 256, 0, stream>>>(sn, snn, MS, D_MODEL);
    // attention (writes attn directly into d_out; final iteration's survives)
    zero_kernel<<<1, 64, 0, stream>>>(colsum, BB * K_SLOTS);
    attn_kernel<<<BB * 3, 256, 0, stream>>>(fn, snn, out_attn, colsum, BB, NPIX, D_MODEL, 3, SCALE);
    updates_kernel<<<BB * K_SLOTS * 3, 256, 0, stream>>>(out_attn, colsum, v, upd, BB, NPIX, D_MODEL);
    // GRU
    gemm(stream, upd, D_MODEL, W_ih, D_MODEL, 1, b_ih, nullptr, gi, 3 * D_MODEL, MS, 3 * D_MODEL, D_MODEL, 0);
    gemm(stream, scur, D_MODEL, W_hh, D_MODEL, 1, b_hh, nullptr, gh, 3 * D_MODEL, MS, 3 * D_MODEL, D_MODEL, 0);
    gru_kernel<<<(MS * D_MODEL + 255) / 256, 256, 0, stream>>>(gi, gh, scur, sbuf, MS, D_MODEL);
    // MLP with residual: slots = s + W2(gelu(W1(LN(s)) + b1)) + b2
    ln_kernel<<<(MS + 7) / 8, 256, 0, stream>>>(sbuf, mg, mbv, tln, nullptr, MS, D_MODEL);
    gemm(stream, tln, D_MODEL, W1, 4 * D_MODEL, 0, b1, nullptr, h1, 4 * D_MODEL, MS, 4 * D_MODEL, D_MODEL, 2 /*gelu*/);
    gemm(stream, h1, 4 * D_MODEL, W2, D_MODEL, 0, b2, sbuf, snxt, D_MODEL, MS, D_MODEL, 4 * D_MODEL, 0);
    float* t = scur; scur = snxt; snxt = t;
  }

  hipMemcpyAsync(out_slots, scur, (size_t)MS * D_MODEL * sizeof(float),
                 hipMemcpyDeviceToDevice, stream);
}
